// DGCNN_54288386621897
// MI455X (gfx1250) — matrix operand order repaired and checked by reading, not compiled, not run on hardware
//
#include <hip/hip_runtime.h>
#include <hip/hip_bf16.h>

typedef _Float16 half_t;
typedef __attribute__((ext_vector_type(16))) _Float16 v16h;
typedef __attribute__((ext_vector_type(8)))  _Float16 v8h;
typedef __attribute__((ext_vector_type(8)))  float    v8f;

#define NGRAPH 4096
#define NPG    64
#define FIN    128
#define HID    32
#define DTOT   97
#define XSTR   128          // padded xcat stride (halves) -> conv1 K padded to 128
#define KTOP   30
#define EPG    1024
#define ETOT   (NGRAPH*EPG)
#define DENSE  352
#define MLPH   128

// Assemble a 16x32 f16 A-fragment for one lane from row-major storage.
// Layout: half e of lane L holds K = 16*(L>>4) + e, M = L&15.
// => 16 contiguous halves at k0, k0 = (L&16)?16:0.
__device__ __forceinline__ v16h make_a(const half_t* p) {
  const v8h lo = *(const v8h*)p;
  const v8h hi = *(const v8h*)(p + 8);
  v16h a;
#pragma unroll
  for (int e = 0; e < 8; ++e) { a[e] = lo[e]; a[e + 8] = hi[e]; }
  return a;
}

// 16 contiguous halves -> v16h (for pre-transposed f16 weights).
__device__ __forceinline__ v16h make_contig(const half_t* p) {
  const v8h lo = *(const v8h*)p;
  const v8h hi = *(const v8h*)(p + 8);
  v16h b;
#pragma unroll
  for (int e = 0; e < 8; ++e) { b[e] = lo[e]; b[e + 8] = hi[e]; }
  return b;
}

// ---------------------------------------------------------------------------
// Kernel W: transpose/convert w_mlp1 [352,128] f32 -> wt [128][352] f16 in ws.
// ---------------------------------------------------------------------------
__global__ __launch_bounds__(256)
void wcvt(const float* __restrict__ wm1, half_t* __restrict__ wt)
{
  int i = blockIdx.x * 256 + threadIdx.x;
  if (i < MLPH * DENSE) {
    int n = i / DENSE, k = i - n * DENSE;
    wt[i] = (half_t)wm1[k * MLPH + n];
  }
}

// ---------------------------------------------------------------------------
// Kernel A: per-graph GCN stack (WMMA) + LDS edge aggregation + sort-pool +
// conv1 (WMMA) / maxpool / conv2 head. Writes head features [B,352] f16 to ws.
// ---------------------------------------------------------------------------
__global__ __launch_bounds__(256)
void gcn_sortpool_head(const float* __restrict__ x,
                       const int*   __restrict__ eidx,
                       const float* __restrict__ W0,  const float* __restrict__ W1c,
                       const float* __restrict__ W2c, const float* __restrict__ W3c,
                       const float* __restrict__ b0,  const float* __restrict__ b1,
                       const float* __restrict__ b2,  const float* __restrict__ b3,
                       const float* __restrict__ wc1, const float* __restrict__ bc1,
                       const float* __restrict__ wc2, const float* __restrict__ bc2,
                       half_t* __restrict__ yws)
{
  __shared__ half_t xh[NPG * FIN];        // graph input, f16        16384 B
  __shared__ half_t xcat_h[NPG * XSTR];   // concat conv outs, f16   16384 B
  __shared__ float  hwb[NPG * HID];       // h @ W (pre-agg)          8192 B
  __shared__ float  agg[NPG * HID];       // aggregated               8192 B
  __shared__ float  dinv_[NPG];
  __shared__ float  scr64[NPG];           // deg, then layer-4 agg
  __shared__ float  keyv[NPG];            // layer-4 hw, then sort key
  __shared__ unsigned char esrc[EPG], edst[EPG];
  __shared__ float  enorm_[EPG];
  __shared__ int    topidx[KTOP];
  __shared__ float  y1[KTOP * 16];
  __shared__ float  yp[15 * 16];
  __shared__ float  lb[4 * 32];           // conv biases

  const int tid  = threadIdx.x;
  const int lane = tid & 31;
  const int wv   = tid >> 5;              // 8 waves
  const int mt   = wv >> 1;               // M tile 0..3
  const int nt   = wv & 1;                // N tile 0..1
  const int row  = mt * 16 + (lane & 15); // A-matrix row for this lane
  const int k0   = (lane & 16) ? 16 : 0;  // A-fragment K offset
  const int bcol = lane & 15;
  const int colg = nt * 16 + bcol;        // output column 0..31
  const int kb   = (lane >> 4) * 16;      // B-fragment K base

  // ---- load weight fragments into VGPRs once per block ----
  // B layout: half e of lane L holds K = kb + e, N = L&15 (per tile).
  v16h w1f[4], w2f, w3f, w4f, wc1f[4];
#pragma unroll
  for (int ks = 0; ks < 4; ++ks)
#pragma unroll
    for (int e = 0; e < 16; ++e)
      w1f[ks][e] = (half_t)W0[(ks * 32 + kb + e) * HID + colg];
#pragma unroll
  for (int e = 0; e < 16; ++e) {
    w2f[e] = (half_t)W1c[(kb + e) * HID + colg];
    w3f[e] = (half_t)W2c[(kb + e) * HID + colg];
    w4f[e] = (nt == 0 && bcol == 0) ? (half_t)W3c[kb + e] : (half_t)0.f;
  }
  // conv1 as GEMM: B = wc1^T, K = feature dim padded 97 -> 128 with zeros.
#pragma unroll
  for (int ks = 0; ks < 4; ++ks)
#pragma unroll
    for (int e = 0; e < 16; ++e) {
      int K = ks * 32 + kb + e;
      wc1f[ks][e] = (K < DTOT) ? (half_t)wc1[bcol * DTOT + K] : (half_t)0.f;
    }
  if (tid < 32) {
    lb[tid]      = b0[tid];
    lb[32 + tid] = b1[tid];
    lb[64 + tid] = b2[tid];
    lb[96 + tid] = (tid == 0) ? b3[0] : 0.f;
  }
  // zero xcat pad columns (96..127) once; col 96 is rewritten per graph.
  for (int i = tid; i < NPG * 32; i += 256) {
    int n = i >> 5, c = 96 + (i & 31);
    xcat_h[n * XSTR + c] = (half_t)0.f;
  }
  __syncthreads();

  for (int g = blockIdx.x; g < NGRAPH; g += gridDim.x) {
    // prefetch next graph's feature block into cache (global_prefetch_b8)
    if (g + gridDim.x < NGRAPH)
      __builtin_prefetch(x + (size_t)(g + gridDim.x) * NPG * FIN + tid * 32, 0, 1);

    // ---- stage X (f32 -> f16) into LDS ----
    const float4* xsrc = (const float4*)(x + (size_t)g * NPG * FIN);
    for (int i = tid; i < NPG * FIN / 4; i += 256) {
      float4 v = xsrc[i];
      xh[i * 4 + 0] = (half_t)v.x; xh[i * 4 + 1] = (half_t)v.y;
      xh[i * 4 + 2] = (half_t)v.z; xh[i * 4 + 3] = (half_t)v.w;
    }
    // ---- edges (local ids) + degree ----
    if (tid < NPG) scr64[tid] = 1.0f;     // self loop
    for (int i = tid; i < EPG; i += 256) {
      esrc[i] = (unsigned char)(eidx[(size_t)g * EPG + i] & 63);
      edst[i] = (unsigned char)(eidx[(size_t)ETOT + (size_t)g * EPG + i] & 63);
    }
    __syncthreads();
    for (int i = tid; i < EPG; i += 256) atomicAdd(&scr64[edst[i]], 1.0f);
    __syncthreads();
    if (tid < NPG) dinv_[tid] = rsqrtf(scr64[tid]);
    __syncthreads();
    for (int i = tid; i < EPG; i += 256) enorm_[i] = dinv_[esrc[i]] * dinv_[edst[i]];

    // ---- GCN layers 0..2 (out dim 32) ----
    for (int l = 0; l < 3; ++l) {
      v8f acc = {};
      if (l == 0) {
#pragma unroll
        for (int ks = 0; ks < 4; ++ks) {
          v16h a = make_a(&xh[row * FIN + ks * 32 + k0]);
          acc = __builtin_amdgcn_wmma_f32_16x16x32_f16(
              false, a, false, w1f[ks], (short)0, acc, false, false);
        }
      } else {
        v16h a  = make_a(&xcat_h[row * XSTR + (l - 1) * HID + k0]);
        v16h wb = (l == 1) ? w2f : w3f;
        acc = __builtin_amdgcn_wmma_f32_16x16x32_f16(
            false, a, false, wb, (short)0, acc, false, false);
      }
      __syncthreads();   // previous-phase LDS readers done before hwb overwrite
#pragma unroll
      for (int i = 0; i < 8; ++i)
        hwb[(mt * 16 + i + 8 * (lane >> 4)) * HID + colg] = acc[i];
      __syncthreads();
      // self-loop contribution
      for (int idx = tid; idx < NPG * HID; idx += 256) {
        int n = idx >> 5;
        agg[idx] = hwb[idx] * dinv_[n] * dinv_[n];
      }
      __syncthreads();
      // edge aggregation: thread (e-lane, feature) -> LDS float atomics
      {
        int f = tid & 31, el = tid >> 5;
        for (int e0 = 0; e0 < EPG; e0 += 8) {
          int e = e0 + el;
          atomicAdd(&agg[edst[e] * HID + f], hwb[esrc[e] * HID + f] * enorm_[e]);
        }
      }
      __syncthreads();
      // tanh + bias -> xcat (f16)
      for (int idx = tid; idx < NPG * HID; idx += 256) {
        int n = idx >> 5, f = idx & 31;
        xcat_h[n * XSTR + l * HID + f] = (half_t)tanhf(agg[idx] + lb[l * 32 + f]);
      }
      __syncthreads();
    }

    // ---- layer 3 (out dim 1, padded to 16-col WMMA) ----
    {
      v16h a = make_a(&xcat_h[row * XSTR + 2 * HID + k0]);
      v8f acc = __builtin_amdgcn_wmma_f32_16x16x32_f16(
          false, a, false, w4f, (short)0, (v8f){}, false, false);
      if (nt == 0 && bcol == 0) {
#pragma unroll
        for (int i = 0; i < 8; ++i)
          keyv[mt * 16 + i + 8 * (lane >> 4)] = acc[i];
      }
    }
    __syncthreads();
    if (tid < NPG) scr64[tid] = keyv[tid] * dinv_[tid] * dinv_[tid];
    __syncthreads();
    for (int e = tid; e < EPG; e += 256)
      atomicAdd(&scr64[edst[e]], keyv[esrc[e]] * enorm_[e]);
    __syncthreads();
    if (tid < NPG) {
      float v = tanhf(scr64[tid] + lb[96]);
      xcat_h[tid * XSTR + 96] = (half_t)v;
      keyv[tid] = v;
    }
    __syncthreads();

    // ---- sort-pool: stable descending rank on keyv (matches argsort(-key)) ----
    if (tid < NPG) {
      float ki = keyv[tid];
      int r = 0;
      for (int j = 0; j < NPG; ++j) {
        float kj = keyv[j];
        r += (kj > ki) || (kj == ki && j < tid);
      }
      if (r < KTOP) topidx[r] = tid;
    }
    __syncthreads();

    // ---- head conv1 as WMMA: [30(pad 32), 97(pad 128)] @ [128, 16] ----
    if (wv < 2) {                          // waves 0,1 = M tiles 0,1
      const int rk   = wv * 16 + (lane & 15);
      const int node = (rk < KTOP) ? topidx[rk] : 0;   // gathered A rows
      v8f acc = {};
#pragma unroll
      for (int ks = 0; ks < 4; ++ks) {
        v16h a = make_a(&xcat_h[node * XSTR + ks * 32 + k0]);
        acc = __builtin_amdgcn_wmma_f32_16x16x32_f16(
            false, a, false, wc1f[ks], (short)0, acc, false, false);
      }
      const float bc = bc1[lane & 15];
#pragma unroll
      for (int i = 0; i < 8; ++i) {
        int r = wv * 16 + i + 8 * (lane >> 4);
        if (r < KTOP) y1[r * 16 + (lane & 15)] = fmaxf(acc[i] + bc, 0.f);
      }
    }
    __syncthreads();
    // ---- maxpool(2) ----
    for (int idx = tid; idx < 15 * 16; idx += 256) {
      int p = idx >> 4, c = idx & 15;
      yp[idx] = fmaxf(y1[(2 * p) * 16 + c], y1[(2 * p + 1) * 16 + c]);
    }
    __syncthreads();
    // ---- conv2 (k=5, 16->32) + relu, flatten as [C2,11] -> ws f16 ----
    for (int idx = tid; idx < 11 * 32; idx += 256) {
      int c2 = idx / 11, p = idx - 11 * c2;
      float s = bc2[c2];
      for (int c1 = 0; c1 < 16; ++c1)
#pragma unroll
        for (int j = 0; j < 5; ++j)
          s += yp[(p + j) * 16 + c1] * wc2[c2 * 80 + c1 * 5 + j];
      yws[(size_t)g * DENSE + idx] = (half_t)fmaxf(s, 0.f);
    }
    __syncthreads();
  }
}

// ---------------------------------------------------------------------------
// Kernel B: [4096,352] @ [352,128] WMMA GEMM + relu + dot with w_mlp2 + bias.
// One block = 16 graphs (one M tile); 8 waves cover the 8 N tiles.
// B-fragments come from the pre-transposed f16 weight (contiguous 32B loads).
// ---------------------------------------------------------------------------
__global__ __launch_bounds__(256)
void mlp_head(const half_t* __restrict__ yws, const half_t* __restrict__ wt,
              const float* __restrict__ bm1,
              const float* __restrict__ wm2, const float* __restrict__ bm2,
              float* __restrict__ out)
{
  __shared__ half_t yt[16 * DENSE];       // 11264 B
  __shared__ float  oacc[16];

  const int tid  = threadIdx.x;
  const int lane = tid & 31;
  const int wv   = tid >> 5;              // N tile 0..7
  const int g0   = blockIdx.x * 16;

  const unsigned int* src  = (const unsigned int*)(yws + (size_t)g0 * DENSE);
  unsigned int*       dstp = (unsigned int*)yt;
  for (int i = tid; i < 16 * DENSE / 2; i += 256) dstp[i] = src[i];
  if (tid < 16) oacc[tid] = 0.f;
  __syncthreads();

  const int col = wv * 16 + (lane & 15);  // 0..127
  const int kb  = (lane >> 4) * 16;
  const int k0  = (lane & 16) ? 16 : 0;
  const int row = lane & 15;

  v8f acc = {};
  for (int ks = 0; ks < DENSE / 32; ++ks) {   // 11 K-steps
    v16h a = make_a(&yt[row * DENSE + ks * 32 + k0]);
    v16h b = make_contig(&wt[(size_t)col * DENSE + ks * 32 + kb]);
    acc = __builtin_amdgcn_wmma_f32_16x16x32_f16(
        false, a, false, b, (short)0, acc, false, false);
  }
  const float bcol = bm1[col];
  const float w2c  = wm2[col];
#pragma unroll
  for (int i = 0; i < 8; ++i) {
    float h = fmaxf(acc[i] + bcol, 0.f);
    atomicAdd(&oacc[i + 8 * (lane >> 4)], h * w2c);
  }
  __syncthreads();
  if (tid < 16) out[g0 + tid] = oacc[tid] + bm2[0];
}

// ---------------------------------------------------------------------------
extern "C" void kernel_launch(void* const* d_in, const int* in_sizes, int n_in,
                              void* d_out, int out_size, void* d_ws, size_t ws_size,
                              hipStream_t stream) {
  (void)in_sizes; (void)n_in; (void)out_size; (void)ws_size;
  const float* x   = (const float*)d_in[0];
  const int*   ei  = (const int*)  d_in[1];   // edge_index [2,E]
  // d_in[2] = batch (uniform 64 nodes/graph -> unused)
  const float* W0  = (const float*)d_in[3];
  const float* W1c = (const float*)d_in[4];
  const float* W2c = (const float*)d_in[5];
  const float* W3c = (const float*)d_in[6];
  const float* b0  = (const float*)d_in[7];
  const float* b1  = (const float*)d_in[8];
  const float* b2  = (const float*)d_in[9];
  const float* b3  = (const float*)d_in[10];
  const float* wc1 = (const float*)d_in[11];
  const float* bc1 = (const float*)d_in[12];
  const float* wc2 = (const float*)d_in[13];
  const float* bc2 = (const float*)d_in[14];
  const float* wm1 = (const float*)d_in[15];
  const float* bm1 = (const float*)d_in[16];
  const float* wm2 = (const float*)d_in[17];
  const float* bm2 = (const float*)d_in[18];

  half_t* yws = (half_t*)d_ws;                       // [4096,352] f16
  half_t* wt  = yws + (size_t)NGRAPH * DENSE;        // [128,352]  f16 (w_mlp1^T)

  wcvt<<<(MLPH * DENSE + 255) / 256, 256, 0, stream>>>(wm1, wt);
  gcn_sortpool_head<<<1024, 256, 0, stream>>>(
      x, ei, W0, W1c, W2c, W3c, b0, b1, b2, b3, wc1, bc1, wc2, bc2, yws);
  mlp_head<<<NGRAPH / 16, 256, 0, stream>>>(
      yws, wt, bm1, wm2, bm2, (float*)d_out);
}
